// CausalSelfAttention_10050223473046
// MI455X (gfx1250) — compile-verified, matrix operand order checked by reading, not run on hardware
//
#include <hip/hip_runtime.h>

typedef __attribute__((ext_vector_type(16))) __bf16 v16bf;
typedef __attribute__((ext_vector_type(8)))  __bf16 v8bf;
typedef __attribute__((ext_vector_type(8)))  float  v8f;

#define T_SEQ  2048
#define NH     16
#define HD     64
#define EMB    1024
#define BATCH  4
#define MROWS  (BATCH * T_SEQ)   // 8192
#define LDSP   40                // padded LDS row (32 data + 8 pad) to dodge bank conflicts

// ---------- helpers ----------

__device__ __forceinline__ __bf16 f2bf(float f) {
  unsigned u = __builtin_bit_cast(unsigned, f);
  u = (u + 0x7FFFu + ((u >> 16) & 1u)) >> 16;   // round-to-nearest-even
  unsigned short h = (unsigned short)u;
  return __builtin_bit_cast(__bf16, h);
}

__device__ __forceinline__ v8f vzero8() {
  v8f z;
#pragma unroll
  for (int i = 0; i < 8; ++i) z[i] = 0.0f;
  return z;
}

// CDNA5 16-bit A/B fragment: elements 0..7 = M[row][klo+0..7], 8..15 = M[row][klo+16..23]
__device__ __forceinline__ v16bf make_frag(const __bf16* p) {
  v8bf lo = *(const v8bf*)(p);
  v8bf hi = *(const v8bf*)(p + 16);
  v16bf r;
#pragma unroll
  for (int i = 0; i < 8; ++i) { r[i] = lo[i]; r[i + 8] = hi[i]; }
  return r;
}

__device__ __forceinline__ v8f wmma_bf16(v16bf a, v16bf b, v8f c) {
  return __builtin_amdgcn_wmma_f32_16x16x32_bf16(
      /*neg_a=*/false, a, /*neg_b=*/false, b,
      /*c_mod=*/(short)0, c, /*reuse_a=*/false, /*reuse_b=*/false);
}

// Async global->LDS copy (CDNA5): writes LDS directly, tracked by ASYNCcnt.
// GV mode: vdst = wave-relative LDS byte offset, vaddr = 64-bit global address.
__device__ __forceinline__ void async_ld_b128(unsigned lds_off, const void* gptr) {
  asm volatile("global_load_async_to_lds_b128 %0, %1, off"
               :: "v"(lds_off), "v"((unsigned long long)(uintptr_t)gptr)
               : "memory");
}
__device__ __forceinline__ void wait_async0() {
  asm volatile("s_wait_asynccnt 0x0" ::: "memory");
}

// ---------- fp32 -> bf16 convert ----------

__global__ void __launch_bounds__(256) cvt_bf16_kernel(const float* __restrict__ in,
                                                       __bf16* __restrict__ out, long n) {
  long i = (long)blockIdx.x * 256 + threadIdx.x;
  if (i < n) out[i] = f2bf(in[i]);
}

// ---------- shared GEMM mainloop ----------
// C[m,n] = sum_k A[m,k] * B[n,k]   (A: MxK row-major, B: NxK row-major -> B^T)
// Block tile 128x128: 8 waves as 4(M) x 2(N), wave tile 32x64, K-step 32.
// Double-buffered async-to-LDS staging; LDS reads stay on the DS path
// (single addrspace(3) base indexed by parity -> ds_load_b128).
// acc[sub][t]: rows = m0 + wm*32 + sub*16 + half*8 + r ; cols = n0 + wn*64 + t*16 + nn

__device__ __forceinline__ void gemm_mainloop(
    const __bf16* __restrict__ A, const __bf16* __restrict__ B, int K,
    long m0, long n0, v8f acc[2][4]) {
  __shared__ __bf16 smem[2][2][128][LDSP];   // [buf][A/B][row][col]

  const int tid  = threadIdx.x;
  const int wave = tid >> 5;
  const int lane = tid & 31;
  const int wm   = wave >> 1;
  const int wn   = wave & 1;
  const int nn   = lane & 15;
  const int klo  = (lane >> 4) << 3;

  // per-thread staging chunks: 512 x (8 bf16) per 128x32 tile
  const int r0  = tid >> 2;
  const int cc0 = (tid & 3) << 3;
  const int r1  = (tid + 256) >> 2;
  const int cc1 = cc0;   // (c & 3) identical for c and c+256

  const int steps = K / 32;

  // prologue: stage K-slab 0 into buffer 0
  async_ld_b128((unsigned)(uintptr_t)&smem[0][0][r0][cc0], &A[(m0 + r0) * (long)K + cc0]);
  async_ld_b128((unsigned)(uintptr_t)&smem[0][1][r0][cc0], &B[(n0 + r0) * (long)K + cc0]);
  async_ld_b128((unsigned)(uintptr_t)&smem[0][0][r1][cc1], &A[(m0 + r1) * (long)K + cc1]);
  async_ld_b128((unsigned)(uintptr_t)&smem[0][1][r1][cc1], &B[(n0 + r1) * (long)K + cc1]);

  for (int i = 0; i < steps; ++i) {
    const int p = i & 1;
    wait_async0();          // this wave's contributions to buf[p] have landed
    __syncthreads();        // publish buf[p]; all reads of buf[1-p] are done
    if (i + 1 < steps) {    // stage next K-slab into buf[1-p]
      const int q  = 1 - p;
      const int kb = (i + 1) * 32;
      async_ld_b128((unsigned)(uintptr_t)&smem[q][0][r0][cc0], &A[(m0 + r0) * (long)K + kb + cc0]);
      async_ld_b128((unsigned)(uintptr_t)&smem[q][1][r0][cc0], &B[(n0 + r0) * (long)K + kb + cc0]);
      async_ld_b128((unsigned)(uintptr_t)&smem[q][0][r1][cc1], &A[(m0 + r1) * (long)K + kb + cc1]);
      async_ld_b128((unsigned)(uintptr_t)&smem[q][1][r1][cc1], &B[(n0 + r1) * (long)K + kb + cc1]);
    }
    if (i + 2 < steps) {    // warm L2 for the slab after next
      __builtin_prefetch(&A[(m0 + (tid >> 1)) * (long)K + (i + 2) * 32], 0, 1);
      __builtin_prefetch(&B[(n0 + (tid >> 1)) * (long)K + (i + 2) * 32], 0, 1);
    }
    // compute from buf[p]: 8 WMMAs, B fragments reused across the two M sub-tiles
    v16bf af0 = make_frag(&smem[p][0][wm * 32 + nn][klo]);
    v16bf af1 = make_frag(&smem[p][0][wm * 32 + 16 + nn][klo]);
#pragma unroll
    for (int t = 0; t < 4; ++t) {
      v16bf bfr = make_frag(&smem[p][1][wn * 64 + t * 16 + nn][klo]);
      acc[0][t] = wmma_bf16(af0, bfr, acc[0][t]);
      acc[1][t] = wmma_bf16(af1, bfr, acc[1][t]);
    }
  }
}

// ---------- QKV projection ----------
// qkv = x @ qkv_w^T + qkv_b ; scatter to Q[B,H,T,D], K[B,H,T,D], Vt[B,H,D,T] (bf16)

__global__ void __launch_bounds__(256) qkv_gemm_kernel(
    const __bf16* __restrict__ X, const __bf16* __restrict__ W,
    const float* __restrict__ bias,
    __bf16* __restrict__ Qo, __bf16* __restrict__ Ko, __bf16* __restrict__ Vt) {
  const long m0 = (long)blockIdx.y * 128;
  const long n0 = (long)blockIdx.x * 128;
  v8f acc[2][4];
#pragma unroll
  for (int s = 0; s < 2; ++s)
#pragma unroll
    for (int t = 0; t < 4; ++t) acc[s][t] = vzero8();
  gemm_mainloop(X, W, EMB, m0, n0, acc);

  const int lane = threadIdx.x & 31;
  const int wave = threadIdx.x >> 5;
  const int wm   = wave >> 1;
  const int wn   = wave & 1;
  const int nn   = lane & 15;
  const int half = lane >> 4;
#pragma unroll
  for (int s = 0; s < 2; ++s)
#pragma unroll
    for (int t = 0; t < 4; ++t) {
      const int n  = (int)n0 + wn * 64 + t * 16 + nn;
      const float bv = bias[n];
#pragma unroll
      for (int r = 0; r < 8; ++r) {
        const int m  = (int)m0 + wm * 32 + s * 16 + half * 8 + r;
        const float v = acc[s][t][r] + bv;
        const int bb = m >> 11;           // /T_SEQ
        const int tt = m & (T_SEQ - 1);
        if (n < EMB) {
          const int hh = n >> 6, d = n & 63;
          Qo[(((long)(bb * NH + hh) * T_SEQ) + tt) * HD + d] = f2bf(v);
        } else if (n < 2 * EMB) {
          const int o2 = n - EMB;
          const int hh = o2 >> 6, d = o2 & 63;
          Ko[(((long)(bb * NH + hh) * T_SEQ) + tt) * HD + d] = f2bf(v);
        } else {
          const int o2 = n - 2 * EMB;
          const int hh = o2 >> 6, d = o2 & 63;
          Vt[(((long)(bb * NH + hh) * HD) + d) * T_SEQ + tt] = f2bf(v);  // V transposed
        }
      }
    }
}

// ---------- flash attention (causal, online softmax) ----------

__global__ void __launch_bounds__(128) attn_kernel(
    const __bf16* __restrict__ Qb, const __bf16* __restrict__ Kb,
    const __bf16* __restrict__ Vt, __bf16* __restrict__ AttO) {
  __shared__ __bf16 Pl[4][16][32];   // per-wave P staging (C-layout -> A-frag)
  const int wave = threadIdx.x >> 5;
  const int lane = threadIdx.x & 31;
  const int nn   = lane & 15;
  const int half = lane >> 4;
  const int klo  = half << 3;
  const int b = blockIdx.z, h = blockIdx.y;
  const int q0 = blockIdx.x * 64 + wave * 16;
  const long bh = (long)(b * NH + h);
  const __bf16* Q  = Qb + bh * T_SEQ * HD;
  const __bf16* Km = Kb + bh * T_SEQ * HD;
  const __bf16* Vm = Vt + bh * HD * T_SEQ;

  // Q fragments for this wave's 16 rows, resident for the whole KV loop
  v16bf qf[2];
#pragma unroll
  for (int ds = 0; ds < 2; ++ds)
    qf[ds] = make_frag(&Q[(q0 + nn) * HD + ds * 32 + klo]);

  v8f o[4];
#pragma unroll
  for (int t = 0; t < 4; ++t) o[t] = vzero8();
  float mi[8], li[8];
#pragma unroll
  for (int r = 0; r < 8; ++r) { mi[r] = -1e30f; li[r] = 0.0f; }

  const float scale = 0.125f;                 // 1/sqrt(64)
  const int nk = blockIdx.x * 2 + 2;          // uniform over block (KV chunks of 32)
  for (int kc = 0; kc < nk; ++kc) {
    const int k0 = kc * 32;
    // S = Q K^T (16x32 as two 16x16 tiles, K-dim = D = 64 -> 2 wmma each)
    v8f s[2];
#pragma unroll
    for (int t = 0; t < 2; ++t) {
      v8f z = vzero8();
      v16bf bk0 = make_frag(&Km[(k0 + t * 16 + nn) * HD + 0 + klo]);
      v16bf bk1 = make_frag(&Km[(k0 + t * 16 + nn) * HD + 32 + klo]);
      z    = wmma_bf16(qf[0], bk0, z);
      s[t] = wmma_bf16(qf[1], bk1, z);
    }
    // online softmax in C/D layout: row m = half*8 + r, col = t*16 + nn
#pragma unroll
    for (int r = 0; r < 8; ++r) {
      const int qg = q0 + half * 8 + r;
      const bool va = (k0 + nn)      <= qg;
      const bool vb = (k0 + 16 + nn) <= qg;
      const float v0 = va ? s[0][r] * scale : -1e30f;
      const float v1 = vb ? s[1][r] * scale : -1e30f;
      float mx = fmaxf(v0, v1);
#pragma unroll
      for (int off = 1; off < 16; off <<= 1)
        mx = fmaxf(mx, __shfl_xor(mx, off, 32));
      const float mn = fmaxf(mi[r], mx);
      const float al = __expf(mi[r] - mn);
      mi[r] = mn;
      const float e0 = va ? __expf(v0 - mn) : 0.0f;
      const float e1 = vb ? __expf(v1 - mn) : 0.0f;
      float rs = e0 + e1;
#pragma unroll
      for (int off = 1; off < 16; off <<= 1)
        rs += __shfl_xor(rs, off, 32);
      li[r] = li[r] * al + rs;
#pragma unroll
      for (int dt = 0; dt < 4; ++dt) o[dt][r] *= al;
      Pl[wave][half * 8 + r][nn]      = f2bf(e0);
      Pl[wave][half * 8 + r][16 + nn] = f2bf(e1);
    }
    // intra-wave LDS producer->consumer ordering
    asm volatile("s_wait_dscnt 0x0" ::: "memory");
    // O += P V  (P 16x32 A-frag from LDS, V from pre-transposed Vt: contiguous rows)
    v16bf pa = make_frag(&Pl[wave][nn][klo]);
#pragma unroll
    for (int dt = 0; dt < 4; ++dt) {
      v16bf bv = make_frag(&Vm[(dt * 16 + nn) * T_SEQ + k0 + klo]);
      o[dt] = wmma_bf16(pa, bv, o[dt]);
    }
  }

  // normalize and emit [B,T,C] bf16 for the projection GEMM
  float inv[8];
#pragma unroll
  for (int r = 0; r < 8; ++r) inv[r] = 1.0f / li[r];
#pragma unroll
  for (int dt = 0; dt < 4; ++dt)
#pragma unroll
    for (int r = 0; r < 8; ++r) {
      const int tq = q0 + half * 8 + r;
      AttO[((long)(b * T_SEQ) + tq) * EMB + h * HD + dt * 16 + nn] =
          f2bf(o[dt][r] * inv[r]);
    }
}

// ---------- output projection ----------

__global__ void __launch_bounds__(256) proj_gemm_kernel(
    const __bf16* __restrict__ A, const __bf16* __restrict__ W,
    const float* __restrict__ bias, float* __restrict__ out) {
  const long m0 = (long)blockIdx.y * 128;
  const long n0 = (long)blockIdx.x * 128;
  v8f acc[2][4];
#pragma unroll
  for (int s = 0; s < 2; ++s)
#pragma unroll
    for (int t = 0; t < 4; ++t) acc[s][t] = vzero8();
  gemm_mainloop(A, W, EMB, m0, n0, acc);

  const int lane = threadIdx.x & 31;
  const int wave = threadIdx.x >> 5;
  const int wm   = wave >> 1;
  const int wn   = wave & 1;
  const int nn   = lane & 15;
  const int half = lane >> 4;
#pragma unroll
  for (int s = 0; s < 2; ++s)
#pragma unroll
    for (int t = 0; t < 4; ++t) {
      const int n = (int)n0 + wn * 64 + t * 16 + nn;
      const float bv = bias[n];
#pragma unroll
      for (int r = 0; r < 8; ++r) {
        const long m = m0 + wm * 32 + s * 16 + half * 8 + r;
        out[m * EMB + n] = acc[s][t][r] + bv;
      }
    }
}

// ---------- launch ----------

extern "C" void kernel_launch(void* const* d_in, const int* in_sizes, int n_in,
                              void* d_out, int out_size, void* d_ws, size_t ws_size,
                              hipStream_t stream) {
  const float* x      = (const float*)d_in[0];
  const float* qkv_w  = (const float*)d_in[1];
  const float* qkv_b  = (const float*)d_in[2];
  const float* proj_w = (const float*)d_in[3];
  const float* proj_b = (const float*)d_in[4];
  float* out = (float*)d_out;

  char* ws = (char*)d_ws;
  size_t off = 0;
  auto take = [&](size_t bytes) -> char* {
    char* p = ws + off;
    off += (bytes + 255) & ~(size_t)255;
    return p;
  };
  __bf16* xb  = (__bf16*)take((size_t)MROWS * EMB * 2);       // 16 MB
  __bf16* wqb = (__bf16*)take((size_t)3 * EMB * EMB * 2);     //  6 MB
  __bf16* wpb = (__bf16*)take((size_t)EMB * EMB * 2);         //  2 MB
  __bf16* Qb  = (__bf16*)take((size_t)MROWS * EMB * 2);       // 16 MB
  __bf16* Kb  = (__bf16*)take((size_t)MROWS * EMB * 2);       // 16 MB
  __bf16* Vtb = (__bf16*)take((size_t)MROWS * EMB * 2);       // 16 MB
  __bf16* Ab  = (__bf16*)take((size_t)MROWS * EMB * 2);       // 16 MB
  (void)ws_size; (void)in_sizes; (void)n_in; (void)out_size;

  // 1) converts
  {
    long n = (long)MROWS * EMB;
    cvt_bf16_kernel<<<dim3((unsigned)((n + 255) / 256)), dim3(256), 0, stream>>>(x, xb, n);
  }
  {
    long n = (long)3 * EMB * EMB;
    cvt_bf16_kernel<<<dim3((unsigned)((n + 255) / 256)), dim3(256), 0, stream>>>(qkv_w, wqb, n);
  }
  {
    long n = (long)EMB * EMB;
    cvt_bf16_kernel<<<dim3((unsigned)((n + 255) / 256)), dim3(256), 0, stream>>>(proj_w, wpb, n);
  }

  // 2) QKV projection: M=8192, N=3072, block tile 128x128
  qkv_gemm_kernel<<<dim3(3 * EMB / 128, MROWS / 128), dim3(256), 0, stream>>>(
      xb, wqb, qkv_b, Qb, Kb, Vtb);

  // 3) flash attention: (T/64, H, B)
  attn_kernel<<<dim3(T_SEQ / 64, NH, BATCH), dim3(128), 0, stream>>>(Qb, Kb, Vtb, Ab);

  // 4) output projection: M=8192, N=1024, fp32 out + bias
  proj_gemm_kernel<<<dim3(EMB / 128, MROWS / 128), dim3(256), 0, stream>>>(
      Ab, wpb, proj_b, out);
}
